// GAT_26199300505907
// MI455X (gfx1250) — matrix operand order repaired and checked by reading, not compiled, not run on hardware
//
#include <hip/hip_runtime.h>

// ---------------------------------------------------------------------------
// GAT link-prediction for MI455X (gfx1250, wave32).
// GEMMs via v_wmma_f32_16x16x32_f16; scatter/softmax via f32 global atomics;
// all reductions via wave32 shuffles.
// ---------------------------------------------------------------------------

typedef __attribute__((ext_vector_type(16))) _Float16 v16h;
typedef __attribute__((ext_vector_type(8)))  _Float16 v8h;
typedef __attribute__((ext_vector_type(8)))  float    v8f;

static constexpr int N_NODES = 50000;
static constexpr int N_EDGES = 800000;
static constexpr int C_IN    = 128;
static constexpr int C_HID   = 64;
static constexpr int N_HEADS = 4;
static constexpr int C_OUT   = 64;
static constexpr int F1      = N_HEADS * C_HID;   // 256
static constexpr int E_TOT   = N_EDGES + N_NODES; // edges + self loops

__device__ __forceinline__ float warp_sum(float v) {
#pragma unroll
  for (int m = 16; m >= 1; m >>= 1) v += __shfl_xor(v, m, 32);
  return v;
}

// float atomic max valid for all sign combinations (init with -FLT_MAX).
__device__ __forceinline__ void atomic_max_f32(float* addr, float val) {
  if (val >= 0.0f) atomicMax((int*)addr, __float_as_int(val));
  else             atomicMin((unsigned int*)addr, __float_as_uint(val));
}

__device__ __forceinline__ void edge_src_dst(const int* __restrict__ es,
                                             const int* __restrict__ ed,
                                             int E, int idx, int& s, int& d) {
  if (idx < E) { s = es[idx]; d = ed[idx]; }
  else         { s = idx - E; d = idx - E; }   // self loop
}

// ------------------------------- utility kernels ---------------------------

__global__ void k_fill(float* __restrict__ p, float v, int n) {
  int i = blockIdx.x * blockDim.x + threadIdx.x;
  if (i < n) p[i] = v;
}

__global__ void k_f32_to_f16(const float* __restrict__ s, _Float16* __restrict__ d, int n) {
  int i = blockIdx.x * blockDim.x + threadIdx.x;
  if (i < n) d[i] = (_Float16)s[i];
}

// W[K,Ncols] f32 -> Wt[Ncols,K] f16 (row per output column -> contiguous K)
__global__ void k_transpose_f16(const float* __restrict__ W, _Float16* __restrict__ Wt,
                                int K, int Ncols) {
  int i = blockIdx.x * blockDim.x + threadIdx.x;
  if (i >= K * Ncols) return;
  int k = i / Ncols, o = i - k * Ncols;
  Wt[(size_t)o * K + k] = (_Float16)W[i];
}

// ------------------------------- WMMA GEMM ---------------------------------
// C[M,Ncols] f32 = A[M,K] f16 (row-major) x Bt[Ncols,K] f16 (transposed B).
// One wave per 16x16 C tile. M%16==0, K%32==0, Ncols%16==0 guaranteed.
__global__ void k_gemm_wmma(const _Float16* __restrict__ A,
                            const _Float16* __restrict__ Bt,
                            float* __restrict__ C, int M, int K, int Ncols) {
  int wave    = (blockIdx.x * blockDim.x + threadIdx.x) >> 5;
  int tiles_n = Ncols >> 4;
  int n_waves = (M >> 4) * tiles_n;
  if (wave >= n_waves) return;                  // wave-uniform: EXEC stays all-1
  int tm = wave / tiles_n, tn = wave - tm * tiles_n;
  int l  = threadIdx.x & 31;
  int hi = l >> 4;                              // lane group (0: lanes 0-15)
  int lm = l & 15;
  const _Float16* arow = A  + (size_t)(tm * 16 + lm) * K;  // A row m
  const _Float16* brow = Bt + (size_t)(tn * 16 + lm) * K;  // B column n
  v8f acc = {};
  for (int kt = 0; kt < K; kt += 32) {
    // A 16x32 layout: elems 0..7 -> K = kt+8*hi.. ; elems 8..15 -> K = kt+16+8*hi..
    v8h a0 = *(const v8h*)(arow + kt + 8 * hi);
    v8h a1 = *(const v8h*)(arow + kt + 16 + 8 * hi);
    // B 32x16 layout: elems 0..7 -> K = kt+8*hi.. ; elems 8..15 -> K = kt+16+8*hi..
    v8h b0 = *(const v8h*)(brow + kt + 8 * hi);
    v8h b1 = *(const v8h*)(brow + kt + 16 + 8 * hi);
    v16h a = __builtin_shufflevector(a0, a1, 0,1,2,3,4,5,6,7,8,9,10,11,12,13,14,15);
    v16h b = __builtin_shufflevector(b0, b1, 0,1,2,3,4,5,6,7,8,9,10,11,12,13,14,15);
    acc = __builtin_amdgcn_wmma_f32_16x16x32_f16(false, a, false, b, (short)0, acc,
                                                 false, false);
  }
  // C/D layout: VGPR r, lane group hi -> M = r + 8*hi, N = lane%16
  float* cp = C + (size_t)(tm * 16 + 8 * hi) * Ncols + tn * 16 + lm;
#pragma unroll
  for (int r = 0; r < 8; ++r) cp[(size_t)r * Ncols] = acc[r];
}

// ------------------------- attention score kernels -------------------------
// a_s[n,h] = dot(h[n,h,:], att_src[h,:]); a_d likewise. One wave per (n,h).
__global__ void k_scores(const float* __restrict__ hf,
                         const float* __restrict__ att_s,
                         const float* __restrict__ att_d,
                         float* __restrict__ a_s, float* __restrict__ a_d,
                         int Nn, int Hh, int C) {
  int wave = (blockIdx.x * blockDim.x + threadIdx.x) >> 5;
  int l = threadIdx.x & 31;
  if (wave >= Nn * Hh) return;
  int n = wave / Hh, h = wave - n * Hh;
  const float* hp  = hf + (size_t)n * Hh * C + (size_t)h * C;
  const float* asp = att_s + h * C;
  const float* adp = att_d + h * C;
  float ss = 0.f, sd = 0.f;
  for (int c = l; c < C; c += 32) { float v = hp[c]; ss += v * asp[c]; sd += v * adp[c]; }
  ss = warp_sum(ss); sd = warp_sum(sd);
  if (l == 0) { a_s[wave] = ss; a_d[wave] = sd; }
}

// --------------------------- edge softmax passes ---------------------------

__global__ void k_edge_max(const int* __restrict__ es, const int* __restrict__ ed,
                           int E, int Etot, const float* __restrict__ a_s,
                           const float* __restrict__ a_d, float* __restrict__ mx, int Hh) {
  int i = blockIdx.x * blockDim.x + threadIdx.x;
  if (i >= Etot) return;
  int s, d; edge_src_dst(es, ed, E, i, s, d);
  for (int h = 0; h < Hh; ++h) {
    float al = a_s[s * Hh + h] + a_d[d * Hh + h];
    al = al > 0.f ? al : 0.2f * al;             // leaky relu 0.2
    atomic_max_f32(&mx[d * Hh + h], al);
  }
}

__global__ void k_edge_sum(const int* __restrict__ es, const int* __restrict__ ed,
                           int E, int Etot, const float* __restrict__ a_s,
                           const float* __restrict__ a_d, const float* __restrict__ mx,
                           float* __restrict__ den, int Hh) {
  int i = blockIdx.x * blockDim.x + threadIdx.x;
  if (i >= Etot) return;
  int s, d; edge_src_dst(es, ed, E, i, s, d);
  for (int h = 0; h < Hh; ++h) {
    float al = a_s[s * Hh + h] + a_d[d * Hh + h];
    al = al > 0.f ? al : 0.2f * al;
    atomicAdd(&den[d * Hh + h], __expf(al - mx[d * Hh + h]));
  }
}

// out[dst,:] += w[head] * feat[src,:]  — one wave per edge, lane-strided feats.
__global__ void k_edge_agg(const int* __restrict__ es, const int* __restrict__ ed,
                           int E, int Etot, const float* __restrict__ a_s,
                           const float* __restrict__ a_d, const float* __restrict__ mx,
                           const float* __restrict__ den, const float* __restrict__ feat,
                           float* __restrict__ outp, int Hh, int C) {
  int wave = (blockIdx.x * blockDim.x + threadIdx.x) >> 5;
  int l = threadIdx.x & 31;
  if (wave >= Etot) return;
  int s, d; edge_src_dst(es, ed, E, wave, s, d);
  float wv = 0.f;
  if (l < Hh) {
    float al = a_s[s * Hh + l] + a_d[d * Hh + l];
    al = al > 0.f ? al : 0.2f * al;
    wv = __expf(al - mx[d * Hh + l]) / (den[d * Hh + l] + 1e-16f);
  }
  int F = Hh * C;
  const float* fs = feat + (size_t)s * F;
  float* od = outp + (size_t)d * F;
  for (int c = l; c < F; c += 32) {
    float wgt = __shfl(wv, c / C, 32);          // broadcast per-head weight
    atomicAdd(&od[c], wgt * fs[c]);
  }
}

// bias (+optional relu) in place; optional f16 copy for next GEMM.
__global__ void k_bias_act(float* __restrict__ io, const float* __restrict__ bias,
                           _Float16* __restrict__ out16, int total, int F, int do_relu) {
  int i = blockIdx.x * blockDim.x + threadIdx.x;
  if (i >= total) return;
  float v = io[i] + bias[i % F];
  if (do_relu) v = fmaxf(v, 0.f);
  io[i] = v;
  if (out16) out16[i] = (_Float16)v;
}

// ------------------------------- decode ------------------------------------
// logits[e] = dot(z[ei0[e]], z[ei1[e]]) over pos then neg. One wave per edge.
__global__ void k_decode(const float* __restrict__ z, const int* __restrict__ pos,
                         const int* __restrict__ neg, int E, float* __restrict__ logits,
                         int C) {
  int wave = (blockIdx.x * blockDim.x + threadIdx.x) >> 5;
  int l = threadIdx.x & 31;
  if (wave >= 2 * E) return;
  const int* ei = (wave < E) ? pos : neg;
  int e = (wave < E) ? wave : wave - E;
  int i = ei[e], j = ei[e + E];
  const float* zi = z + (size_t)i * C;
  const float* zj = z + (size_t)j * C;
  float acc = 0.f;
  for (int c = l; c < C; c += 32) acc += zi[c] * zj[c];
  acc = warp_sum(acc);
  if (l == 0) logits[wave] = acc;
}

// ------------------------------- launcher ----------------------------------

static inline dim3 g1(long long threads) { return dim3((unsigned)((threads + 255) / 256)); }

extern "C" void kernel_launch(void* const* d_in, const int* in_sizes, int n_in,
                              void* d_out, int out_size, void* d_ws, size_t ws_size,
                              hipStream_t stream) {
  const float* x    = (const float*)d_in[0];
  const int*   pos  = (const int*)  d_in[1];
  const int*   neg  = (const int*)  d_in[2];
  const float* W1   = (const float*)d_in[3];
  const float* asW1 = (const float*)d_in[4];
  const float* adW1 = (const float*)d_in[5];
  const float* b1   = (const float*)d_in[6];
  const float* W2   = (const float*)d_in[7];
  const float* asW2 = (const float*)d_in[8];
  const float* adW2 = (const float*)d_in[9];
  const float* b2   = (const float*)d_in[10];
  float* logits = (float*)d_out;

  // ---- workspace layout (256B aligned slabs) ----
  char* ws = (char*)d_ws;
  size_t off = 0;
  auto alloc = [&](size_t bytes) {
    size_t o = off; off = (off + bytes + 255) & ~(size_t)255; return o;
  };
  float*    h1   = (float*)   (ws + alloc((size_t)N_NODES * F1 * 4));   // 51.2 MB
  float*    agg1 = (float*)   (ws + alloc((size_t)N_NODES * F1 * 4));   // 51.2 MB
  _Float16* xh   = (_Float16*)(ws + alloc((size_t)N_NODES * C_IN * 2)); // 12.8 MB
  _Float16* z1h  = (_Float16*)(ws + alloc((size_t)N_NODES * F1 * 2));   // 25.6 MB
  _Float16* w1t  = (_Float16*)(ws + alloc((size_t)F1 * C_IN * 2));
  _Float16* w2t  = (_Float16*)(ws + alloc((size_t)C_OUT * F1 * 2));
  float*    as1  = (float*)(ws + alloc((size_t)N_NODES * N_HEADS * 4));
  float*    ad1  = (float*)(ws + alloc((size_t)N_NODES * N_HEADS * 4));
  float*    m1   = (float*)(ws + alloc((size_t)N_NODES * N_HEADS * 4));
  float*    dn1  = (float*)(ws + alloc((size_t)N_NODES * N_HEADS * 4));
  float*    as2  = (float*)(ws + alloc((size_t)N_NODES * 4));
  float*    ad2  = (float*)(ws + alloc((size_t)N_NODES * 4));
  float*    m2   = (float*)(ws + alloc((size_t)N_NODES * 4));
  float*    dn2  = (float*)(ws + alloc((size_t)N_NODES * 4));
  // Reuse dead slabs: h1 is dead after conv1 aggregation; xh dead after GEMM1.
  float* h2   = h1;          // N*64*4 = 12.8 MB  <= 51.2 MB
  float* agg2 = (float*)xh;  // N*64*4 = 12.8 MB  == xh size

  const int* pos_src = pos;            // edge_index[0] = message source j
  const int* pos_dst = pos + N_EDGES;  // edge_index[1] = aggregation target i
  const float NEG_INF = -3.402823466e+38f;

  // ---- precision conversion / weight transpose ----
  k_f32_to_f16   <<<g1((long long)N_NODES * C_IN), 256, 0, stream>>>(x, xh, N_NODES * C_IN);
  k_transpose_f16<<<g1(C_IN * F1),  256, 0, stream>>>(W1, w1t, C_IN, F1);
  k_transpose_f16<<<g1(F1 * C_OUT), 256, 0, stream>>>(W2, w2t, F1, C_OUT);

  // ---- conv1: h1 = x @ W1 (WMMA) ----
  k_gemm_wmma<<<g1((long long)(N_NODES / 16) * (F1 / 16) * 32), 256, 0, stream>>>(
      xh, w1t, h1, N_NODES, C_IN, F1);
  k_scores<<<g1((long long)N_NODES * N_HEADS * 32), 256, 0, stream>>>(
      h1, asW1, adW1, as1, ad1, N_NODES, N_HEADS, C_HID);
  k_fill<<<g1(N_NODES * N_HEADS), 256, 0, stream>>>(m1, NEG_INF, N_NODES * N_HEADS);
  k_fill<<<g1(N_NODES * N_HEADS), 256, 0, stream>>>(dn1, 0.f, N_NODES * N_HEADS);
  k_fill<<<g1((long long)N_NODES * F1), 256, 0, stream>>>(agg1, 0.f, N_NODES * F1);
  k_edge_max<<<g1(E_TOT), 256, 0, stream>>>(pos_src, pos_dst, N_EDGES, E_TOT,
                                            as1, ad1, m1, N_HEADS);
  k_edge_sum<<<g1(E_TOT), 256, 0, stream>>>(pos_src, pos_dst, N_EDGES, E_TOT,
                                            as1, ad1, m1, dn1, N_HEADS);
  k_edge_agg<<<g1((long long)E_TOT * 32), 256, 0, stream>>>(
      pos_src, pos_dst, N_EDGES, E_TOT, as1, ad1, m1, dn1, h1, agg1, N_HEADS, C_HID);
  k_bias_act<<<g1((long long)N_NODES * F1), 256, 0, stream>>>(
      agg1, b1, z1h, N_NODES * F1, F1, /*relu=*/1);

  // ---- conv2: h2 = z1 @ W2 (WMMA), heads=1 ----
  k_gemm_wmma<<<g1((long long)(N_NODES / 16) * (C_OUT / 16) * 32), 256, 0, stream>>>(
      z1h, w2t, h2, N_NODES, F1, C_OUT);
  k_scores<<<g1((long long)N_NODES * 32), 256, 0, stream>>>(
      h2, asW2, adW2, as2, ad2, N_NODES, 1, C_OUT);
  k_fill<<<g1(N_NODES), 256, 0, stream>>>(m2, NEG_INF, N_NODES);
  k_fill<<<g1(N_NODES), 256, 0, stream>>>(dn2, 0.f, N_NODES);
  k_fill<<<g1((long long)N_NODES * C_OUT), 256, 0, stream>>>(agg2, 0.f, N_NODES * C_OUT);
  k_edge_max<<<g1(E_TOT), 256, 0, stream>>>(pos_src, pos_dst, N_EDGES, E_TOT,
                                            as2, ad2, m2, 1);
  k_edge_sum<<<g1(E_TOT), 256, 0, stream>>>(pos_src, pos_dst, N_EDGES, E_TOT,
                                            as2, ad2, m2, dn2, 1);
  k_edge_agg<<<g1((long long)E_TOT * 32), 256, 0, stream>>>(
      pos_src, pos_dst, N_EDGES, E_TOT, as2, ad2, m2, dn2, h2, agg2, 1, C_OUT);
  k_bias_act<<<g1((long long)N_NODES * C_OUT), 256, 0, stream>>>(
      agg2, b2, (_Float16*)nullptr, N_NODES * C_OUT, C_OUT, /*relu=*/0);

  // ---- decode: 2E edge dot products ----
  k_decode<<<g1((long long)2 * N_EDGES * 32), 256, 0, stream>>>(
      agg2, pos, neg, N_EDGES, logits, C_OUT);
}
